// Transformer_31396210933928
// MI455X (gfx1250) — compile-verified, matrix operand order checked by reading, not run on hardware
//
#include <hip/hip_runtime.h>

// ---------------------------------------------------------------------------
// MI455X (gfx1250) fused window-transformer block.
// wave32, WMMA bf16 16x16x32 (f32 accumulate), async global->LDS weight
// staging when the toolchain exposes the gfx1250 async builtins.
// ---------------------------------------------------------------------------

#define DEV static __device__ __forceinline__

typedef __attribute__((ext_vector_type(16))) __bf16 v16bf;
typedef __attribute__((ext_vector_type(8)))  float  v8f;
typedef int v4i_t __attribute__((vector_size(4 * sizeof(int))));

#define BATCH   8
#define DIMC    256
#define HIMG    128
#define WIMG    128
#define WS      15
#define NTOK    225        // WS*WS
#define MP      240        // tokens padded to multiple of 16
#define NHEADS  8
#define HD      32         // head dim
#define SCALE   0.1767766952966369f
#define EPSF    1e-6f

// bf16 weight mirror sizes (elements)
#define WQN (768 * 256)
#define W1N (1024 * 256)
#define W2N (256 * 1024)

// LDS strides (elements)
#define XWS 264            // 256 + 8 bf16 pad (row = 528B, 16B aligned)
#define QRS 34             // f32 rows for raw q/k
#define VBS 40
#define KVS 40
#define HMS 1040           // 1024 + 16
#define W2S 136            // 128 + 8

// ---- attention kernel LDS layout (bytes) ----
#define A_XW_OFF 0                      // [240][264] bf16 = 126720
#define A_WT_OFF 126720                 // [ 96][264] bf16 =  50688
#define A_QR_OFF 177408                 // [240][ 34] f32  =  32640
#define A_KR_OFF 210048                 // [240][ 34] f32  =  32640
#define A_VB_OFF 242688                 // [256][ 40] bf16 =  20480
#define A_KT_OFF 263168                 // [ 32][264] bf16 =  16896
#define A_KV_OFF 280064                 // [ 32][ 40] bf16 =   2560
#define A_KS_OFF 282624                 // [ 32] f32       =    128
#define A_NM_OFF 282752                 // [240] f32       =    960
#define SMEM_ATTN 283712

// ---- mlp kernel LDS layout (bytes) ----
#define M_XA_OFF 0                      // [ 64][ 264] bf16 =  33792
#define M_W1_OFF 33792                  // [128][ 264] bf16 =  67584
#define M_HM_OFF 101376                 // [ 64][1040] bf16 = 133120
#define M_W2_OFF 234496                 // [256][ 136] bf16 =  69632
#define SMEM_MLP 304128

// ---------------- async global->LDS (gfx1250) ----------------
#if defined(__HIP_DEVICE_COMPILE__) && \
    __has_builtin(__builtin_amdgcn_global_load_async_to_lds_b128)
#define USE_ASYNC_LDS 1
#else
#define USE_ASYNC_LDS 0
#endif

DEV void cp16(const __bf16* g, __bf16* l) {
#if USE_ASYNC_LDS
  __builtin_amdgcn_global_load_async_to_lds_b128(
      (__attribute__((address_space(1))) v4i_t*)(unsigned long long)g,
      (__attribute__((address_space(3))) v4i_t*)(unsigned)(unsigned long long)l,
      0, 0);
#else
  *(uint4*)l = *(const uint4*)g;
#endif
}
DEV void wait_async() {
#if USE_ASYNC_LDS
#if __has_builtin(__builtin_amdgcn_s_wait_asynccnt)
  __builtin_amdgcn_s_wait_asynccnt(0);
#else
  asm volatile("s_wait_asynccnt 0x0" ::: "memory");
#endif
#endif
}

// ---------------- scalar helpers ----------------
DEV __bf16 f2bf(float f) {
  unsigned u = __builtin_bit_cast(unsigned, f);
  unsigned short r = (unsigned short)((u + 0x7FFFu + ((u >> 16) & 1u)) >> 16);
  return __builtin_bit_cast(__bf16, r);
}
DEV float bf2f(__bf16 b) {
  unsigned short s = __builtin_bit_cast(unsigned short, b);
  unsigned u = ((unsigned)s) << 16;
  return __builtin_bit_cast(float, u);
}
DEV float softplusf(float x) { return (x > 20.f) ? x : log1pf(expf(x)); }

DEV v8f wmma_bf16(v16bf a, v16bf b, v8f c) {
  return __builtin_amdgcn_wmma_f32_16x16x32_bf16(false, a, false, b,
                                                 (short)0, c, false, false);
}

// A/B operand stored [row][K] (row = M for A, N for B-transposed-weights).
// CDNA5 16-bit operand striping: lanes 0-15 hold K = kk+[0..7],[16..23];
// lanes 16-31 hold K = kk+[8..15],[24..31]; row = lane&15 (+tile base).
DEV v16bf load_rk_bf(const __bf16* p0, int rowBase, int stride, int kk, int lane) {
  const int hi = (lane >> 4) & 1;
  const __bf16* p = p0 + (size_t)(rowBase + (lane & 15)) * stride + kk + hi * 8;
  v16bf a;
#pragma unroll
  for (int i = 0; i < 8; ++i) { a[i] = p[i]; a[i + 8] = p[i + 16]; }
  return a;
}
DEV v16bf load_rk_f32(const float* p0, int rowBase, int stride, int kk, int lane) {
  const int hi = (lane >> 4) & 1;
  const float* p = p0 + (size_t)(rowBase + (lane & 15)) * stride + kk + hi * 8;
  v16bf a;
#pragma unroll
  for (int i = 0; i < 8; ++i) { a[i] = f2bf(p[i]); a[i + 8] = f2bf(p[i + 16]); }
  return a;
}
// B operand stored in natural [K][N] order (used for V matrix, KV matrix).
DEV v16bf load_b_kn(const __bf16* p0, int kBase, int stride, int nBase, int lane) {
  const int hi = (lane >> 4) & 1;
  const int n = nBase + (lane & 15);
  v16bf b;
#pragma unroll
  for (int i = 0; i < 16; ++i) {
    int kOff = ((i < 8) ? i : i + 8) + hi * 8;
    b[i] = p0[(size_t)(kBase + kOff) * stride + n];
  }
  return b;
}

// ---------------------------------------------------------------------------
// Kernel 0: convert the three GEMM weight matrices to bf16 mirrors in ws.
// ---------------------------------------------------------------------------
__global__ void cvt_weights_kernel(const float* __restrict__ wqkv,
                                   const float* __restrict__ wfc1,
                                   const float* __restrict__ wfc2,
                                   __bf16* __restrict__ wqb,
                                   __bf16* __restrict__ w1b,
                                   __bf16* __restrict__ w2b) {
  const int i = blockIdx.x * 256 + threadIdx.x;   // grid covers 262144
  if (i < WQN) wqb[i] = f2bf(wqkv[i]);
  if (i < W1N) { w1b[i] = f2bf(wfc1[i]); w2b[i] = f2bf(wfc2[i]); }
}

// ---------------------------------------------------------------------------
// Kernel 1: depthwise 3x3 conv + bias + residual, NCHW -> NHWC shortcut xs
// ---------------------------------------------------------------------------
__global__ void conv_pos_kernel(const float* __restrict__ x,
                                const float* __restrict__ cw,
                                const float* __restrict__ cb,
                                float* __restrict__ xs) {
  const int n = blockIdx.x * 256 + threadIdx.x;     // 8*256*128*128 threads
  const int w  = n & 127;
  const int h  = (n >> 7) & 127;
  const int c  = (n >> 14) & 255;
  const int b  = n >> 22;
  const size_t plane = (((size_t)b * DIMC + c) * HIMG) * WIMG;
  float s = x[plane + (size_t)h * WIMG + w] + cb[c];
  const float* wp = cw + c * 9;
#pragma unroll
  for (int dy = 0; dy < 3; ++dy) {
    const int yy = h + dy - 1;
    if (yy < 0 || yy >= HIMG) continue;
#pragma unroll
    for (int dx = 0; dx < 3; ++dx) {
      const int xx = w + dx - 1;
      if (xx < 0 || xx >= WIMG) continue;
      s += x[plane + (size_t)yy * WIMG + xx] * wp[dy * 3 + dx];
    }
  }
  xs[((((size_t)b * HIMG + h) * WIMG) + w) * DIMC + c] = s;
}

// ---------------------------------------------------------------------------
// Kernel 2: per-window linear attention + shortcut residual -> y1 (NHWC f32)
// One workgroup (8 waves) per window; 648 windows.
// ---------------------------------------------------------------------------
__global__ void attn_kernel(const float* __restrict__ xs,
                            const __bf16* __restrict__ wqb,
                            const float* __restrict__ bqkv,
                            float* __restrict__ y1) {
  extern __shared__ unsigned char smem[];
  __bf16* XW = (__bf16*)(smem + A_XW_OFF);
  __bf16* WT = (__bf16*)(smem + A_WT_OFF);
  float*  QR = (float*) (smem + A_QR_OFF);
  float*  KR = (float*) (smem + A_KR_OFF);
  __bf16* VB = (__bf16*)(smem + A_VB_OFF);
  __bf16* KT = (__bf16*)(smem + A_KT_OFF);
  __bf16* KV = (__bf16*)(smem + A_KV_OFF);
  float*  KS = (float*) (smem + A_KS_OFF);
  float*  NM = (float*) (smem + A_NM_OFF);

  const int tid  = threadIdx.x;
  const int lane = tid & 31;
  const int wv   = tid >> 5;

  const int win = blockIdx.x;                // 0..647
  const int b   = win / 81;
  const int rem = win % 81;
  const int oh  = (rem / 9) * WS;
  const int ow  = (rem % 9) * WS;

  // ---- stage window tokens (zero-padded like the reference) as bf16 ----
  for (int idx = tid; idx < MP * DIMC; idx += 256) {
    const int t = idx >> 8, c = idx & 255;
    float v = 0.f;
    if (t < NTOK) {
      const int gh = oh + t / WS, gw = ow + t % WS;
      if (gh < HIMG && gw < WIMG)
        v = xs[((((size_t)b * HIMG + gh) * WIMG) + gw) * DIMC + c];
    }
    XW[t * XWS + c] = f2bf(v);
  }
  __syncthreads();

  for (int h = 0; h < NHEADS; ++h) {
    // ---- async-stage 96 bf16 weight rows (q,k,v slices of this head) ----
    for (int idx = tid; idx < 96 * 32; idx += 256) {  // 16B chunks
      const int jj = idx >> 5, c8 = (idx & 31) * 8;
      const int j = (jj >> 5) * DIMC + h * HD + (jj & 31);
      cp16(wqb + (size_t)j * DIMC + c8, WT + jj * XWS + c8);
    }
    wait_async();
    __syncthreads();

    // ---- qkv GEMM: [240x256] @ [256x96] (q|k|v of this head) ----
    for (int t = wv; t < 15 * 6; t += 8) {
      const int mt = t / 6, nt = t % 6;
      v8f acc = {};
#pragma unroll
      for (int kk = 0; kk < DIMC; kk += 32)
        acc = wmma_bf16(load_rk_bf(XW, mt * 16, XWS, kk, lane),
                        load_rk_bf(WT, nt * 16, XWS, kk, lane), acc);
      const int col = nt * 16 + (lane & 15);    // 0..95
      const int sel = col >> 5;                  // 0=q 1=k 2=v
      const int d   = col & 31;
      const float bias = bqkv[sel * 256 + h * HD + d];
      const int rb = mt * 16 + ((lane >> 4) << 3);
#pragma unroll
      for (int r = 0; r < 8; ++r) {
        const float val = acc[r] + bias;
        const int row = rb + r;
        if (sel == 0)      QR[row * QRS + d] = val;
        else if (sel == 1) KR[row * QRS + d] = val;
        else               VB[row * VBS + d] = f2bf(val);
      }
    }
    __syncthreads();

    // ---- per-token l2norm + softplus; build k^T bf16; zero K padding ----
    {
      const int n = tid;                        // 0..255 exactly
      if (n < NTOK) {
        float q[HD], k[HD], sq = 0.f, sk = 0.f;
#pragma unroll
        for (int d = 0; d < HD; ++d) {
          q[d] = QR[n * QRS + d]; sq += q[d] * q[d];
          k[d] = KR[n * QRS + d]; sk += k[d] * k[d];
        }
        const float dq = fmaxf(sqrtf(sq), 1e-12f);
        const float dk = fmaxf(sqrtf(sk), 1e-12f);
#pragma unroll
        for (int d = 0; d < HD; ++d) {
          QR[n * QRS + d]  = softplusf(q[d] / dq);
          KT[d * XWS + n]  = f2bf(softplusf(k[d] / dk));
        }
      } else {                                   // pad tokens 225..255
        const __bf16 z = f2bf(0.f);
#pragma unroll
        for (int d = 0; d < HD; ++d) {
          KT[d * XWS + n] = z;
          VB[n * VBS + d] = z;
          if (n < MP) QR[n * QRS + d] = 0.f;
        }
      }
    }
    __syncthreads();

    // ---- kv = SCALE * k^T @ v  [32x256]@[256x32]  (waves 0-3) ----
    // ---- ksum[d] = sum_n k[n][d] + EPS           (wave 4)     ----
    if (wv < 4) {
      const int dt = wv >> 1, et = wv & 1;
      v8f acc = {};
#pragma unroll
      for (int kk = 0; kk < 256; kk += 32)
        acc = wmma_bf16(load_rk_bf(KT, dt * 16, XWS, kk, lane),
                        load_b_kn(VB, kk, VBS, et * 16, lane), acc);
      const int e  = et * 16 + (lane & 15);
      const int rb = dt * 16 + ((lane >> 4) << 3);
#pragma unroll
      for (int r = 0; r < 8; ++r)
        KV[(rb + r) * KVS + e] = f2bf(acc[r] * SCALE);
    } else if (wv == 4) {
      const int d = lane;
      float s = 0.f;
      for (int n = 0; n < NTOK; ++n) s += bf2f(KT[d * XWS + n]);
      KS[d] = s + EPSF;
    }
    __syncthreads();

    // ---- norm[n] = 1 / (q[n] . ksum) ----
    if (tid < MP) {
      const int n = tid;
      float nv = 0.f;
      if (n < NTOK) {
        float s = 0.f;
#pragma unroll
        for (int d = 0; d < HD; ++d) s += QR[n * QRS + d] * KS[d];
        nv = 1.f / s;
      }
      NM[n] = nv;
    }
    __syncthreads();

    // ---- out = (q @ kv) * norm ; y1 = out + shortcut ----
    for (int t = wv; t < 15 * 2; t += 8) {
      const int mt = t >> 1, et = t & 1;
      v8f acc = {};
      acc = wmma_bf16(load_rk_f32(QR, mt * 16, QRS, 0, lane),
                      load_b_kn(KV, 0, KVS, et * 16, lane), acc);
      const int e   = et * 16 + (lane & 15);
      const int cch = h * HD + e;
      const int rb  = mt * 16 + ((lane >> 4) << 3);
#pragma unroll
      for (int r = 0; r < 8; ++r) {
        const int n = rb + r;
        if (n < NTOK) {
          const int gh = oh + n / WS, gw = ow + n % WS;
          if (gh < HIMG && gw < WIMG) {
            const size_t off =
                ((((size_t)b * HIMG + gh) * WIMG) + gw) * DIMC + cch;
            y1[off] = acc[r] * NM[n] + xs[off];
          }
        }
      }
    }
    __syncthreads();
  }
}

// ---------------------------------------------------------------------------
// Kernel 3: MLP (256->1024->256, no act) + residual + NHWC->NCHW store.
// 64-token tiles; FC2 accumulators persist in registers across K chunks.
// ---------------------------------------------------------------------------
__global__ void mlp_kernel(const float* __restrict__ y1,
                           const __bf16* __restrict__ w1b,
                           const float* __restrict__ bfc1,
                           const __bf16* __restrict__ w2b,
                           const float* __restrict__ bfc2,
                           float* __restrict__ out) {
  extern __shared__ unsigned char smem[];
  __bf16* XA = (__bf16*)(smem + M_XA_OFF);
  __bf16* W1 = (__bf16*)(smem + M_W1_OFF);
  __bf16* HM = (__bf16*)(smem + M_HM_OFF);
  __bf16* W2 = (__bf16*)(smem + M_W2_OFF);

  const int tid  = threadIdx.x;
  const int lane = tid & 31;
  const int wv   = tid >> 5;
  const size_t m0 = (size_t)blockIdx.x * 64;

  // stage 64 activation rows as bf16
  for (int idx = tid; idx < 64 * DIMC; idx += 256) {
    const int r = idx >> 8, c = idx & 255;
    XA[r * XWS + c] = f2bf(y1[(m0 + r) * DIMC + c]);
  }
  __syncthreads();

  // ---- FC1: hmid = y1 @ wfc1^T + bfc1, 128-column chunks ----
  for (int jt = 0; jt < 8; ++jt) {
    for (int idx = tid; idx < 128 * 32; idx += 256) {   // 16B chunks
      const int r = idx >> 5, c8 = (idx & 31) * 8;
      cp16(w1b + (size_t)(jt * 128 + r) * DIMC + c8, W1 + r * XWS + c8);
    }
    wait_async();
    __syncthreads();
    for (int t = wv; t < 32; t += 8) {
      const int mt = t >> 3, nt = t & 7;
      v8f acc = {};
#pragma unroll
      for (int kk = 0; kk < DIMC; kk += 32)
        acc = wmma_bf16(load_rk_bf(XA, mt * 16, XWS, kk, lane),
                        load_rk_bf(W1, nt * 16, XWS, kk, lane), acc);
      const int jcol = jt * 128 + nt * 16 + (lane & 15);
      const float bias = bfc1[jcol];
      const int rb = mt * 16 + ((lane >> 4) << 3);
#pragma unroll
      for (int r = 0; r < 8; ++r)
        HM[(rb + r) * HMS + jcol] = f2bf(acc[r] + bias);
    }
    __syncthreads();
  }

  // ---- FC2: out = hmid @ wfc2^T, K chunked 128; persistent accumulators ----
  v8f acc[8] = {};
  for (int kt = 0; kt < 8; ++kt) {
    for (int idx = tid; idx < 256 * 16; idx += 256) {   // 16B chunks
      const int r = idx >> 4, c8 = (idx & 15) * 8;
      cp16(w2b + (size_t)r * 1024 + kt * 128 + c8, W2 + r * W2S + c8);
    }
    wait_async();
    __syncthreads();
    int ti = 0;
    for (int t = wv; t < 64; t += 8, ++ti) {
      const int mt = t >> 4, nt = t & 15;
#pragma unroll
      for (int kk = 0; kk < 128; kk += 32)
        acc[ti] = wmma_bf16(load_rk_bf(HM, mt * 16, HMS, kt * 128 + kk, lane),
                            load_rk_bf(W2, nt * 16, W2S, kk, lane), acc[ti]);
    }
    __syncthreads();
  }

  // ---- bias + residual + NHWC->NCHW store ----
  int ti = 0;
  for (int t = wv; t < 64; t += 8, ++ti) {
    const int mt = t >> 4, nt = t & 15;
    const int c = nt * 16 + (lane & 15);
    const float b2 = bfc2[c];
    const int rb = mt * 16 + ((lane >> 4) << 3);
#pragma unroll
    for (int r = 0; r < 8; ++r) {
      const size_t tok = m0 + rb + r;
      const float val = acc[ti][r] + b2 + y1[tok * DIMC + c];
      const size_t bb = tok >> 14;            // /(128*128)
      const size_t rm = tok & 16383;
      out[(bb * DIMC + (size_t)c) * 16384 + rm] = val;
    }
  }
}

// ---------------------------------------------------------------------------
extern "C" void kernel_launch(void* const* d_in, const int* in_sizes, int n_in,
                              void* d_out, int out_size, void* d_ws, size_t ws_size,
                              hipStream_t stream) {
  const float* x      = (const float*)d_in[0];
  const float* conv_w = (const float*)d_in[1];
  const float* conv_b = (const float*)d_in[2];
  const float* wqkv   = (const float*)d_in[3];
  const float* bqkv   = (const float*)d_in[4];
  const float* wfc1   = (const float*)d_in[5];
  const float* bfc1   = (const float*)d_in[6];
  const float* wfc2   = (const float*)d_in[7];
  const float* bfc2   = (const float*)d_in[8];
  float* outp = (float*)d_out;

  const size_t nelem = (size_t)BATCH * DIMC * HIMG * WIMG;  // 33,554,432
  float*  xs  = (float*)d_ws;          // NHWC shortcut after conv stage
  float*  y1  = xs + nelem;            // NHWC after attention + residual
  __bf16* wqb = (__bf16*)(y1 + nelem); // bf16 weight mirrors
  __bf16* w1b = wqb + WQN;
  __bf16* w2b = w1b + W1N;

  (void)hipFuncSetAttribute((const void*)attn_kernel,
                            hipFuncAttributeMaxDynamicSharedMemorySize,
                            SMEM_ATTN);
  (void)hipFuncSetAttribute((const void*)mlp_kernel,
                            hipFuncAttributeMaxDynamicSharedMemorySize,
                            SMEM_MLP);

  cvt_weights_kernel<<<dim3(W1N / 256), dim3(256), 0, stream>>>(
      wqkv, wfc1, wfc2, wqb, w1b, w2b);

  conv_pos_kernel<<<dim3((unsigned)(nelem / 256)), dim3(256), 0, stream>>>(
      x, conv_w, conv_b, xs);

  attn_kernel<<<dim3(648), dim3(256), SMEM_ATTN, stream>>>(xs, wqb, bqkv, y1);

  mlp_kernel<<<dim3((unsigned)(nelem / DIMC / 64)), dim3(256), SMEM_MLP,
               stream>>>(y1, w1b, bfc1, w2b, bfc2, outp);

  (void)in_sizes; (void)n_in; (void)out_size; (void)ws_size;
}
